// Encoder_49246095016204
// MI455X (gfx1250) — matrix operand order repaired and checked by reading, not compiled, not run on hardware
//
#include <hip/hip_runtime.h>
#include <hip/hip_bf16.h>

typedef __bf16 bf16;
typedef __attribute__((ext_vector_type(16))) __bf16 v16bf;
typedef __attribute__((ext_vector_type(8)))  __bf16 v8bf;
typedef __attribute__((ext_vector_type(8)))  float   v8f;

#define BN_S 0.99950037f   // 1/sqrt(1+1e-3)

// ---------------------------------------------------------------------------
// WMMA helpers (CDNA5 16x16x32 bf16 -> f32)
// ---------------------------------------------------------------------------
__device__ __forceinline__ v8f wmma_bf16(v16bf a, v16bf b, v8f c) {
  return __builtin_amdgcn_wmma_f32_16x16x32_bf16(false, a, false, b, (short)0, c,
                                                 false, false);
}

// A fragment (16x32, 16-bit): lane<16 -> K {0..7,16..23}, lane>=16 -> +8
__device__ __forceinline__ v16bf load_frag_a(const bf16* tile, int stride, int k0,
                                             int lane) {
  int row = lane & 15;
  int hi  = lane >> 4;
  const bf16* q = tile + row * stride + k0 + hi * 8;
  v8bf lo = *(const v8bf*)q;
  v8bf hh = *(const v8bf*)(q + 16);
  v16bf f;
#pragma unroll
  for (int i = 0; i < 8; ++i) { f[i] = lo[i]; f[i + 8] = hh[i]; }
  return f;
}

// B fragment (32x16, 16-bit): lane = column; lane<16 K {0..7,16..23}, lane>=16 -> +8
// (tile stored transposed: rows = N, cols = K)
__device__ __forceinline__ v16bf load_frag_b(const bf16* tile, int stride, int k0,
                                             int lane) {
  int row = lane & 15;
  int hi  = lane >> 4;
  const bf16* q = tile + row * stride + k0 + hi * 8;
  v8bf lo = *(const v8bf*)q;
  v8bf hh = *(const v8bf*)(q + 16);
  v16bf f;
#pragma unroll
  for (int i = 0; i < 8; ++i) { f[i] = lo[i]; f[i + 8] = hh[i]; }
  return f;
}

// ---------------------------------------------------------------------------
// Embedding gather + f32 -> bf16 cast.  grid = 8192 rows, block = 256
// ---------------------------------------------------------------------------
__global__ void embed_cast(const int* __restrict__ x, const float* __restrict__ E,
                           bf16* __restrict__ Xe) {
  int row = blockIdx.x;
  int id  = x[row];
  const float* src = E + (size_t)id * 512;
  bf16* dst = Xe + (size_t)row * 512;
  for (int c = threadIdx.x; c < 512; c += 256) dst[c] = (bf16)src[c];
}

// ---------------------------------------------------------------------------
// Weight transpose + cast: W[K][N] f32 -> WT[N][K] bf16
// ---------------------------------------------------------------------------
__global__ void wtrans(const float* __restrict__ W, bf16* __restrict__ WT, int K,
                       int N) {
  long idx   = (long)blockIdx.x * 256 + threadIdx.x;
  long total = (long)K * N;
  if (idx < total) {
    long k = idx / N, n = idx % N;
    WT[n * (long)K + k] = (bf16)W[idx];
  }
}

// ---------------------------------------------------------------------------
// h0 -> bf16 staging buffer (step -1 ping-pong slot).  h0l: [2][32][512]
// hbuf: [dir][parity][32][512].  grid = 128, block = 256
// ---------------------------------------------------------------------------
__global__ void hinit(const float* __restrict__ h0l, bf16* __restrict__ hbuf) {
  int idx = blockIdx.x * 256 + threadIdx.x;
  if (idx < 2 * 32 * 512) {
    int dir  = idx >> 14;
    int rest = idx & 16383;
    hbuf[(dir * 2 + 1) * 16384 + rest] = (bf16)h0l[idx];
  }
}

// ---------------------------------------------------------------------------
// Input-projection GEMM:  XG[8192][1536] = X[8192][K] @ W[K][1536] + bias
// W supplied transposed as WT[1536][K] bf16.  grid = (64,12), block = 256.
// ---------------------------------------------------------------------------
__global__ void gemm_xg(const bf16* __restrict__ X, const bf16* __restrict__ WT,
                        const float* __restrict__ bias, float* __restrict__ XG,
                        int K) {
  __shared__ bf16 As[128 * 40];
  __shared__ bf16 Bs[128 * 40];
  int tid = threadIdx.x, lane = tid & 31, wid = tid >> 5;
  int m0 = blockIdx.x * 128;
  int n0 = blockIdx.y * 128;
  int p  = wid & 3;   // m-tile pair -> m-tiles {2p, 2p+1}
  int q  = wid >> 2;  // n half      -> n-tiles {4q..4q+3}
  v8f acc[2][4] = {};

  for (int k0 = 0; k0 < K; k0 += 32) {
    __syncthreads();
#pragma unroll
    for (int c = tid; c < 512; c += 256) {  // 128 rows x 4 x 16B chunks
      int row = c >> 2, cc = c & 3;
      *(uint4*)&As[row * 40 + cc * 8] =
          *(const uint4*)&X[(size_t)(m0 + row) * K + k0 + cc * 8];
      *(uint4*)&Bs[row * 40 + cc * 8] =
          *(const uint4*)&WT[(size_t)(n0 + row) * K + k0 + cc * 8];
    }
    __syncthreads();
    v16bf af[2], bfm[4];
#pragma unroll
    for (int i = 0; i < 2; ++i) af[i] = load_frag_a(&As[(p * 2 + i) * 16 * 40], 40, 0, lane);
#pragma unroll
    for (int j = 0; j < 4; ++j) bfm[j] = load_frag_b(&Bs[(q * 4 + j) * 16 * 40], 40, 0, lane);
#pragma unroll
    for (int i = 0; i < 2; ++i)
#pragma unroll
      for (int j = 0; j < 4; ++j) acc[i][j] = wmma_bf16(af[i], bfm[j], acc[i][j]);
  }

  int hi = lane >> 4, ln = lane & 15;
#pragma unroll
  for (int i = 0; i < 2; ++i)
#pragma unroll
    for (int j = 0; j < 4; ++j) {
      int nc   = n0 + (q * 4 + j) * 16 + ln;
      float bv = bias[nc];
#pragma unroll
      for (int e = 0; e < 8; ++e) {
        int row = m0 + (p * 2 + i) * 16 + e + hi * 8;
        XG[(size_t)row * 1536 + nc] = acc[i][j][e] + bv;
      }
    }
}

// ---------------------------------------------------------------------------
// Persistent recurrent scan.  grid = 16 WGs (8 per direction), block = 256.
// Each WG owns 64 h-columns (-> 192 gate columns), keeps its Wh slab in LDS
// for all 256 steps, h state in f32 registers, syncs WGs per step via
// per-step arrive counters + agent-scope fences.
// LDS: whS[192][520] bf16 + hA[32][520] bf16  = 232,960 B
// ---------------------------------------------------------------------------
__global__ void gru_scan(const bf16* __restrict__ whT0, const bf16* __restrict__ whT1,
                         const float* __restrict__ xg0, const float* __restrict__ xg1,
                         const float* __restrict__ br0, const float* __restrict__ br1,
                         const float* __restrict__ h0,   // [2][32][512] this layer
                         bf16* __restrict__ hbuf,        // [2][2][32][512]
                         bf16* __restrict__ outb,        // layer0 out (scaled bf16)
                         float* __restrict__ outf,       // final out (scaled f32)
                         float* __restrict__ hidout,     // [2][32][512] scaled
                         int* __restrict__ counters,     // [2][256]
                         int finalLayer) {
  extern __shared__ bf16 lds[];
  bf16* whS = lds;                 // [192][520]
  bf16* hA  = lds + 192 * 520;     // [32][520]

  int tid = threadIdx.x, lane = tid & 31, wid = tid >> 5;
  int dir = blockIdx.x >> 3;
  int wg  = blockIdx.x & 7;
  const bf16*  whT  = dir ? whT1 : whT0;
  const float* xg   = dir ? xg1 : xg0;
  const float* brec = dir ? br1 : br0;
  int*  cnt = counters + dir * 256;
  bf16* hb  = hbuf + (size_t)dir * 2 * 32 * 512;

  // one-time Wh slab stage: rows 0..63=z, 64..127=r, 128..191=h gate cols
  for (int c = tid; c < 192 * 64; c += 256) {  // 192 rows x 64 x 16B chunks
    int row  = c >> 6, cc = c & 63;
    int grow = (row >> 6) * 512 + wg * 64 + (row & 63);
    *(uint4*)&whS[row * 520 + cc * 8] =
        *(const uint4*)&whT[(size_t)grow * 512 + cc * 8];
  }

  int g = wid >> 1, mt = wid & 1;
  int hi = lane >> 4, ln = lane & 15;
  int col = wg * 64 + g * 16 + ln;  // h column 0..511

  v8f hprev;
#pragma unroll
  for (int e = 0; e < 8; ++e) {
    int b = mt * 16 + e + hi * 8;
    hprev[e] = h0[(size_t)dir * 32 * 512 + b * 512 + col];
  }
  __syncthreads();

  for (int s = 0; s < 256; ++s) {
    if (s > 0) {
      if (tid == 0) {
        while (__hip_atomic_load(&cnt[s - 1], __ATOMIC_RELAXED,
                                 __HIP_MEMORY_SCOPE_AGENT) < 8)
          __builtin_amdgcn_s_sleep(1);
      }
      __syncthreads();
      __builtin_amdgcn_fence(__ATOMIC_ACQUIRE, "agent");
    }
    // stage h_{s-1} (bf16, 32x512) -> LDS
    const bf16* hsrc = hb + ((s + 1) & 1) * 16384;
    for (int c = tid; c < 2048; c += 256) {
      int row = c >> 6, cc = c & 63;
      *(uint4*)&hA[row * 520 + cc * 8] = *(const uint4*)&hsrc[row * 512 + cc * 8];
    }
    __syncthreads();

    v8f cz = {}, cr = {}, ch = {};
#pragma unroll 2
    for (int k0 = 0; k0 < 512; k0 += 32) {
      v16bf a  = load_frag_a(&hA[mt * 16 * 520], 520, k0, lane);
      v16bf bz = load_frag_b(&whS[(g * 16) * 520], 520, k0, lane);
      v16bf br = load_frag_b(&whS[(64 + g * 16) * 520], 520, k0, lane);
      v16bf bh = load_frag_b(&whS[(128 + g * 16) * 520], 520, k0, lane);
      cz = wmma_bf16(a, bz, cz);
      cr = wmma_bf16(a, br, cr);
      ch = wmma_bf16(a, bh, ch);
    }

    int t = dir ? (255 - s) : s;
    bf16* hdst = hb + (s & 1) * 16384;
    float bz = brec[col], brr = brec[512 + col], bhh = brec[1024 + col];
#pragma unroll
    for (int e = 0; e < 8; ++e) {
      int b = mt * 16 + e + hi * 8;
      size_t xb = ((size_t)b * 256 + t) * 1536 + col;
      float xz = xg[xb], xr = xg[xb + 512], xh = xg[xb + 1024];
      float z  = 1.f / (1.f + __expf(-(xz + cz[e] + bz)));
      float r  = 1.f / (1.f + __expf(-(xr + cr[e] + brr)));
      float hh = tanhf(xh + r * (ch[e] + bhh));
      float hn = z * hprev[e] + (1.f - z) * hh;
      hprev[e] = hn;
      hdst[b * 512 + col] = (bf16)hn;
      size_t ob = ((size_t)b * 256 + t) * 1024 + dir * 512 + col;
      if (finalLayer) outf[ob] = hn * BN_S;
      else            outb[ob] = (bf16)(hn * BN_S);
    }
    __builtin_amdgcn_fence(__ATOMIC_RELEASE, "agent");
    __syncthreads();
    if (tid == 0)
      __hip_atomic_fetch_add(&cnt[s], 1, __ATOMIC_RELAXED, __HIP_MEMORY_SCOPE_AGENT);
  }

#pragma unroll
  for (int e = 0; e < 8; ++e) {
    int b = mt * 16 + e + hi * 8;
    hidout[(size_t)dir * 32 * 512 + b * 512 + col] = hprev[e] * BN_S;
  }
}

// ---------------------------------------------------------------------------
extern "C" void kernel_launch(void* const* d_in, const int* in_sizes, int n_in,
                              void* d_out, int out_size, void* d_ws, size_t ws_size,
                              hipStream_t stream) {
  (void)in_sizes; (void)n_in; (void)out_size; (void)ws_size;
  const int*   x    = (const int*)d_in[0];
  const float* h0   = (const float*)d_in[1];   // [2][2][32][512]
  const float* E    = (const float*)d_in[2];
  const float* Wx0f = (const float*)d_in[3];
  const float* Wh0f = (const float*)d_in[4];
  const float* b0f  = (const float*)d_in[5];   // [2][1536]
  const float* Wx0b = (const float*)d_in[6];
  const float* Wh0b = (const float*)d_in[7];
  const float* b0b  = (const float*)d_in[8];
  const float* Wx1f = (const float*)d_in[9];
  const float* Wh1f = (const float*)d_in[10];
  const float* b1f  = (const float*)d_in[11];
  const float* Wx1b = (const float*)d_in[12];
  const float* Wh1b = (const float*)d_in[13];
  const float* b1b  = (const float*)d_in[14];
  float* out = (float*)d_out;

  size_t off = 0;
  char* base = (char*)d_ws;
  auto alloc = [&](size_t bytes) -> char* {
    char* p = base + off;
    off += (bytes + 255) & ~(size_t)255;
    return p;
  };
  bf16* wxt0f = (bf16*)alloc(1536 * 512 * 2);
  bf16* wht0f = (bf16*)alloc(1536 * 512 * 2);
  bf16* wxt0b = (bf16*)alloc(1536 * 512 * 2);
  bf16* wht0b = (bf16*)alloc(1536 * 512 * 2);
  bf16* wxt1f = (bf16*)alloc(1536 * 1024 * 2);
  bf16* wht1f = (bf16*)alloc(1536 * 512 * 2);
  bf16* wxt1b = (bf16*)alloc(1536 * 1024 * 2);
  bf16* wht1b = (bf16*)alloc(1536 * 512 * 2);
  bf16* xemb  = (bf16*)alloc((size_t)8192 * 512 * 2);
  bf16* out0  = (bf16*)alloc((size_t)8192 * 1024 * 2);
  float* xgf  = (float*)alloc((size_t)8192 * 1536 * 4);
  float* xgb  = (float*)alloc((size_t)8192 * 1536 * 4);
  bf16* hbuf  = (bf16*)alloc(2 * 2 * 32 * 512 * 2);
  int*  cnt0  = (int*)alloc(2 * 256 * 4);
  int*  cnt1  = (int*)alloc(2 * 256 * 4);

  const size_t SMEM = (192 * 520 + 32 * 520) * sizeof(bf16);  // 232,960 B
  float* hidbase = out + (size_t)8192 * 1024;

  (void)hipMemsetAsync(cnt0, 0, 2 * 2 * 256 * 4, stream);  // cnt0+cnt1 contiguous

  embed_cast<<<8192, 256, 0, stream>>>(x, E, xemb);
  wtrans<<<(512 * 1536 + 255) / 256, 256, 0, stream>>>(Wx0f, wxt0f, 512, 1536);
  wtrans<<<(512 * 1536 + 255) / 256, 256, 0, stream>>>(Wh0f, wht0f, 512, 1536);
  wtrans<<<(512 * 1536 + 255) / 256, 256, 0, stream>>>(Wx0b, wxt0b, 512, 1536);
  wtrans<<<(512 * 1536 + 255) / 256, 256, 0, stream>>>(Wh0b, wht0b, 512, 1536);
  wtrans<<<(1024 * 1536 + 255) / 256, 256, 0, stream>>>(Wx1f, wxt1f, 1024, 1536);
  wtrans<<<(512 * 1536 + 255) / 256, 256, 0, stream>>>(Wh1f, wht1f, 512, 1536);
  wtrans<<<(1024 * 1536 + 255) / 256, 256, 0, stream>>>(Wx1b, wxt1b, 1024, 1536);
  wtrans<<<(512 * 1536 + 255) / 256, 256, 0, stream>>>(Wh1b, wht1b, 512, 1536);

  // ---- layer 0 ----
  gemm_xg<<<dim3(64, 12), 256, 0, stream>>>(xemb, wxt0f, b0f, xgf, 512);
  gemm_xg<<<dim3(64, 12), 256, 0, stream>>>(xemb, wxt0b, b0b, xgb, 512);
  hinit<<<128, 256, 0, stream>>>(h0, hbuf);
  gru_scan<<<16, 256, SMEM, stream>>>(wht0f, wht0b, xgf, xgb, b0f + 1536,
                                      b0b + 1536, h0, hbuf, out0, out, hidbase,
                                      cnt0, 0);
  // ---- layer 1 ----
  gemm_xg<<<dim3(64, 12), 256, 0, stream>>>(out0, wxt1f, b1f, xgf, 1024);
  gemm_xg<<<dim3(64, 12), 256, 0, stream>>>(out0, wxt1b, b1b, xgb, 1024);
  hinit<<<128, 256, 0, stream>>>(h0 + 2 * 32 * 512, hbuf);
  gru_scan<<<16, 256, SMEM, stream>>>(wht1f, wht1b, xgf, xgb, b1f + 1536,
                                      b1b + 1536, h0 + 2 * 32 * 512, hbuf, out0,
                                      out, hidbase + 2 * 32 * 512, cnt1, 1);
}
